// MultiHeadAttention_32074815767300
// MI455X (gfx1250) — matrix operand order repaired and checked by reading, not compiled, not run on hardware
//
#include <hip/hip_runtime.h>

// ---------------------------------------------------------------------------
// MHA forward for MI455X (gfx1250, wave32, WMMA + Tensor Data Mover).
// ~103 GFLOP vs ~0.2 GB HBM traffic -> compute bound. All three GEMM stages
// run on v_wmma_f32_16x16x32_f16 (f16 in, f32 accumulate); GEMM tiles are
// staged into LDS by the TDM (tensor_load_to_lds, double-buffered, padded
// rows), attention is flash-style (score tensor never materialized).
// ---------------------------------------------------------------------------

typedef __attribute__((ext_vector_type(16))) _Float16 v16h;
typedef __attribute__((ext_vector_type(8)))  float    v8f;
typedef __attribute__((ext_vector_type(4)))  unsigned int v4u;
typedef __attribute__((ext_vector_type(8)))  int v8i;
typedef __attribute__((ext_vector_type(4)))  int v4i;

#define DEVINL __device__ __forceinline__

constexpr int BATCH = 8, SEQ = 1024, DIM = 1024, HEADS = 16, HDIM = 64;
constexpr int MTOT = BATCH * SEQ;   // 8192 GEMM rows
constexpr int NQKV = 3 * DIM;       // 3072
constexpr int LSTR = 40;            // LDS tile row stride in halves (32 + 8 pad)

DEVINL v8f wmma_f16(v16h a, v16h b, v8f c) {
  return __builtin_amdgcn_wmma_f32_16x16x32_f16(false, a, false, b,
                                                (short)0, c, false, false);
}

// A-fragment 16x32 (row-major source, stride in halves).
// Lanes 0-15 row m hold K {0..15}, lanes 16-31 hold K {16..31}.
DEVINL v16h frag_a(const _Float16* tile, int stride, int lane) {
  int m = lane & 15, ks = lane >> 4;
  const _Float16* p = tile + m * stride + ks * 16;
  v16h a;
#pragma unroll
  for (int i = 0; i < 8; ++i) { a[i] = p[i]; a[i + 8] = p[8 + i]; }
  return a;
}

// B-fragment 32x16: lane holds column n=(lane&15); elements e=0..15 are
// K = (lane>=16 ? 16 : 0)+e  -> 16 contiguous halves from a K-major source.
DEVINL v16h frag_b(const _Float16* col_base) {
  v16h b;
#pragma unroll
  for (int i = 0; i < 16; ++i) b[i] = col_base[i];
  return b;
}

// ---------------------------------------------------------------------------
// TDM: async 2D f16 tile (tile_w x tile_h) from global into LDS with row
// padding.  Row = tile_w halves; pad_interval code 3 = every 16 DWORDs (64 B,
// one 32-half row), pad_amount code 3 = 4 DWORDs (8 halves) -> LSTR = 40.
// D# packing per CDNA5 ISA ch.8 (group0: flags/lds/global/type, group1:
// sizes/strides, groups 2-3 zero for 2D tiles).
// ---------------------------------------------------------------------------
DEVINL void tdm_load_2d_f16(unsigned lds_off, const _Float16* gptr,
                            unsigned tensor_w, unsigned tensor_h,
                            unsigned tile_w, unsigned tile_h) {
  unsigned long long ga = (unsigned long long)(uintptr_t)gptr;
  v4u g0;
  g0[0] = 1u;                                          // count=1, user D#
  g0[1] = lds_off;                                     // LDS byte address
  g0[2] = (unsigned)(ga & 0xffffffffu);                // global_addr[31:0]
  g0[3] = (unsigned)((ga >> 32) & 0x01ffffffu)         // global_addr[56:32]
          | (2u << 30);                                // type = 2 ("image")
  v8i g1;
  unsigned d0 = (1u << 16)                             // data_size = 2 bytes
              | (1u << 20)                             // pad_enable
              | (3u << 22)                             // pad_interval: 16 DW
              | (3u << 25);                            // pad_amount: 4 DW
  g1[0] = (int)d0;
  g1[1] = (int)(tensor_w << 16);                       // tensor_dim0[15:0]
  g1[2] = (int)((tensor_w >> 16) | (tensor_h << 16));  // dim0 hi | dim1 lo
  g1[3] = (int)((tensor_h >> 16) | (tile_w << 16));    // dim1 hi | tile_dim0
  g1[4] = (int)(tile_h & 0xffffu);                     // tile_dim1 (dim2 = 0)
  g1[5] = (int)tensor_w;                               // dim0_stride[31:0]
  g1[6] = 0;                                           // stride hi, dim1_stride lo
  g1[7] = 0;
  v4i z = {0, 0, 0, 0};
#if __clang_major__ >= 23
  v8i z8 = {0, 0, 0, 0, 0, 0, 0, 0};
  __builtin_amdgcn_tensor_load_to_lds(g0, g1, z, z, z8, 0);
#else
  __builtin_amdgcn_tensor_load_to_lds(g0, g1, z, z, 0);
#endif
}

// ---------------------------------------------------------------------------
// Preprocessing: x f32 -> f16; weights f32 -> f16 transposed ([N][K]) so GEMM
// B-tiles are K-contiguous 2D tiles a TDM descriptor can fetch directly.
// ---------------------------------------------------------------------------
__global__ __launch_bounds__(256) void cvt_f16_kernel(
    const float* __restrict__ src, _Float16* __restrict__ dst) {
  size_t i = (size_t)blockIdx.x * 256 + threadIdx.x;
  dst[i] = (_Float16)src[i];
}

__global__ __launch_bounds__(256) void transpose_cvt_kernel(
    const float* __restrict__ W, _Float16* __restrict__ WT, int K, int N) {
  __shared__ _Float16 t[32][33];
  int bx = blockIdx.x * 32, by = blockIdx.y * 32;   // col (N) / row (K) base
  int lx = threadIdx.x & 31, ly = threadIdx.x >> 5;
#pragma unroll
  for (int r = ly; r < 32; r += 8)
    t[r][lx] = (_Float16)W[(size_t)(by + r) * N + bx + lx];
  __syncthreads();
#pragma unroll
  for (int r = ly; r < 32; r += 8)
    WT[(size_t)(bx + r) * K + by + lx] = t[lx][r];
}

// ---------------------------------------------------------------------------
// TDM-staged, double-buffered 128x128 WMMA GEMM core (8 waves: each 32x64).
// A: [M][K] f16 row-major.  BT: [N][K] f16 (pre-transposed).  Wave 0 issues
// the next tile pair while everyone computes on the current one.
// ---------------------------------------------------------------------------
struct GemmAcc { v8f a[2][4]; };

DEVINL void gemm_core(const _Float16* __restrict__ A, const _Float16* __restrict__ BT,
                      _Float16 (*As)[128 * LSTR], _Float16 (*Bs)[128 * LSTR],
                      int mb, int nb, int kdim, int tid, GemmAcc& acc) {
  const int wave = tid >> 5, lane = tid & 31;
  const int mloc = (wave >> 1) * 32, nloc = (wave & 1) * 64;
  const bool issuer = (tid < 32);

  if (issuer) {
    tdm_load_2d_f16((unsigned)(uintptr_t)&As[0][0], A + (size_t)mb * kdim,
                    kdim, MTOT, 32, 128);
    tdm_load_2d_f16((unsigned)(uintptr_t)&Bs[0][0], BT + (size_t)nb * kdim,
                    kdim, 0x10000u, 32, 128);
  }
  int it = 0;
  for (int k0 = 0; k0 < kdim; k0 += 32, ++it) {
    const int cur = it & 1;
    const bool more = (k0 + 32 < kdim);
    if (more && issuer) {
      tdm_load_2d_f16((unsigned)(uintptr_t)&As[cur ^ 1][0],
                      A + (size_t)mb * kdim + k0 + 32, kdim, MTOT, 32, 128);
      tdm_load_2d_f16((unsigned)(uintptr_t)&Bs[cur ^ 1][0],
                      BT + (size_t)nb * kdim + k0 + 32, kdim, 0x10000u, 32, 128);
    }
    if (more) __builtin_amdgcn_s_wait_tensorcnt((short)2);  // current pair done
    else      __builtin_amdgcn_s_wait_tensorcnt((short)0);
    __syncthreads();                    // LDS tile visible to all 8 waves

    v16h af[2];
#pragma unroll
    for (int mi = 0; mi < 2; ++mi)
      af[mi] = frag_a(&As[cur][0] + (mloc + mi * 16) * LSTR, LSTR, lane);
#pragma unroll
    for (int ni = 0; ni < 4; ++ni) {
      v16h bf = frag_b(&Bs[cur][0] + (nloc + ni * 16 + (lane & 15)) * LSTR
                       + (lane >> 4) * 16);
#pragma unroll
      for (int mi = 0; mi < 2; ++mi)
        acc.a[mi][ni] = wmma_f16(af[mi], bf, acc.a[mi][ni]);
    }
    __syncthreads();                    // done reading before TDM reuses buffer
  }
}

// ---------------------------------------------------------------------------
// Kernel 1: fused QKV projection.  Scatters Q,K as [B,H,N,64] and V
// transposed as [B,H,64,N] (so attention's V B-fragments are contiguous).
// ---------------------------------------------------------------------------
__global__ __launch_bounds__(256) void qkv_gemm_kernel(
    const _Float16* __restrict__ x16, const _Float16* __restrict__ WqkvT,
    const float* __restrict__ bqkv,
    _Float16* __restrict__ Qo, _Float16* __restrict__ Ko,
    _Float16* __restrict__ Vt) {
  __shared__ _Float16 As[2][128 * LSTR];
  __shared__ _Float16 Bs[2][128 * LSTR];
  const int tid = threadIdx.x, wave = tid >> 5, lane = tid & 31;
  const int nb = blockIdx.x * 128, mb = blockIdx.y * 128;
  const int mloc = (wave >> 1) * 32, nloc = (wave & 1) * 64;

  GemmAcc acc;
#pragma unroll
  for (int mi = 0; mi < 2; ++mi)
#pragma unroll
    for (int ni = 0; ni < 4; ++ni) { v8f z = {}; acc.a[mi][ni] = z; }

  gemm_core(x16, WqkvT, As, Bs, mb, nb, DIM, tid, acc);

#pragma unroll
  for (int mi = 0; mi < 2; ++mi)
#pragma unroll
    for (int ni = 0; ni < 4; ++ni)
#pragma unroll
      for (int v = 0; v < 8; ++v) {
        int row = mb + mloc + mi * 16 + v + (lane >> 4) * 8;
        int col = nb + nloc + ni * 16 + (lane & 15);
        _Float16 hv = (_Float16)(acc.a[mi][ni][v] + bqkv[col]);
        int part = col >> 10, jj = col & 1023, h = jj >> 6, d = jj & 63;
        int b = row >> 10, n = row & 1023;
        size_t rm = (((size_t)(b * HEADS + h) * SEQ) + n) * HDIM + d;
        if (part == 0)      Qo[rm] = hv;
        else if (part == 1) Ko[rm] = hv;
        else                Vt[(((size_t)(b * HEADS + h) * HDIM) + d) * SEQ + n] = hv;
      }
}

// ---------------------------------------------------------------------------
// Kernel 2: flash attention.  Block = (b, h, 128 query rows); wave owns 16
// rows, Q fragments live in registers across the whole key loop.  Per 32-key
// tile: 4 WMMA for S = Q K^T, online softmax via 16-lane shfl_xor reductions,
// P bounced through per-wave LDS (C-layout -> A-layout), 4 WMMA for O += P V.
// Next key tile prefetched into cache (global_prefetch).
// ---------------------------------------------------------------------------
__global__ __launch_bounds__(256) void flash_attn_kernel(
    const _Float16* __restrict__ Qi, const _Float16* __restrict__ Ki,
    const _Float16* __restrict__ Vt, _Float16* __restrict__ ctx) {
  constexpr int PSTR = 40;
  __shared__ _Float16 Ps[8 * 16 * PSTR];

  const int tid = threadIdx.x, wave = tid >> 5, lane = tid & 31;
  const int qb = blockIdx.x & 7, bh = blockIdx.x >> 3;
  const int h = bh & 15, b = bh >> 4;
  const size_t hoff = (size_t)bh * SEQ * HDIM;
  const size_t voff = (size_t)bh * HDIM * SEQ;
  const int qrow = qb * 128 + wave * 16;

  v16h qf[2];
  {
    const _Float16* qp = Qi + hoff + (size_t)(qrow + (lane & 15)) * HDIM;
    int ks = lane >> 4;
#pragma unroll
    for (int kc = 0; kc < 2; ++kc) {
      int base = kc * 32 + ks * 16;
#pragma unroll
      for (int i = 0; i < 8; ++i) {
        qf[kc][i] = qp[base + i];
        qf[kc][8 + i] = qp[base + 8 + i];
      }
    }
  }

  v8f o[4];
#pragma unroll
  for (int t = 0; t < 4; ++t) { v8f z = {}; o[t] = z; }
  float mrun[8], lrun[8];
#pragma unroll
  for (int v = 0; v < 8; ++v) { mrun[v] = -3.0e38f; lrun[v] = 0.0f; }

  _Float16* pw = Ps + wave * 16 * PSTR;

  for (int kb = 0; kb < SEQ; kb += 32) {
    if (kb + 32 < SEQ) {   // pull next key/value tile toward the WGP
      __builtin_prefetch(Ki + hoff + (size_t)(kb + 32 + (lane & 31)) * HDIM, 0, 0);
      __builtin_prefetch(Vt + voff + (size_t)(lane & 31) * SEQ + kb + 32, 0, 0);
    }
    v8f s0 = {}, s1 = {};
#pragma unroll
    for (int kc = 0; kc < 2; ++kc) {
      int kbase = kc * 32 + (lane >> 4) * 16;
      v16h bf0 = frag_b(Ki + hoff + (size_t)(kb + (lane & 15)) * HDIM + kbase);
      s0 = wmma_f16(qf[kc], bf0, s0);
      v16h bf1 = frag_b(Ki + hoff + (size_t)(kb + 16 + (lane & 15)) * HDIM + kbase);
      s1 = wmma_f16(qf[kc], bf1, s1);
    }

    float alpha[8];
#pragma unroll
    for (int v = 0; v < 8; ++v) {
      float a0 = s0[v] * 0.125f;               // 1/sqrt(64)
      float a1 = s1[v] * 0.125f;
      float mx = fmaxf(a0, a1);
      mx = fmaxf(mx, __shfl_xor(mx, 1));
      mx = fmaxf(mx, __shfl_xor(mx, 2));
      mx = fmaxf(mx, __shfl_xor(mx, 4));
      mx = fmaxf(mx, __shfl_xor(mx, 8));
      float mnew = fmaxf(mrun[v], mx);
      alpha[v] = __expf(mrun[v] - mnew);
      float p0 = __expf(a0 - mnew), p1 = __expf(a1 - mnew);
      float rs = p0 + p1;
      rs += __shfl_xor(rs, 1); rs += __shfl_xor(rs, 2);
      rs += __shfl_xor(rs, 4); rs += __shfl_xor(rs, 8);
      lrun[v] = lrun[v] * alpha[v] + rs;
      mrun[v] = mnew;
      int prow = v + (lane >> 4) * 8, pcol = lane & 15;
      pw[prow * PSTR + pcol]      = (_Float16)p0;
      pw[prow * PSTR + pcol + 16] = (_Float16)p1;
    }
#pragma unroll
    for (int t = 0; t < 4; ++t)
#pragma unroll
      for (int v = 0; v < 8; ++v) o[t][v] *= alpha[v];

    asm volatile("s_wait_dscnt 0" ::: "memory");   // P stores -> P loads (same wave)
    v16h pf = frag_a(pw, PSTR, lane);
#pragma unroll
    for (int t = 0; t < 4; ++t) {
      v16h vf = frag_b(Vt + voff + (size_t)(t * 16 + (lane & 15)) * SEQ
                       + kb + (lane >> 4) * 16);
      o[t] = wmma_f16(pf, vf, o[t]);
    }
  }

#pragma unroll
  for (int v = 0; v < 8; ++v) {
    float inv = 1.0f / lrun[v];
    int row = qrow + v + (lane >> 4) * 8;
#pragma unroll
    for (int t = 0; t < 4; ++t) {
      int d = t * 16 + (lane & 15);
      ctx[((size_t)b * SEQ + row) * DIM + h * HDIM + d] = (_Float16)(o[t][v] * inv);
    }
  }
}

// ---------------------------------------------------------------------------
// Kernel 3: output projection.  ctx(f16) x WoT(f16) + bo -> f32 out.
// ---------------------------------------------------------------------------
__global__ __launch_bounds__(256) void out_gemm_kernel(
    const _Float16* __restrict__ ctx, const _Float16* __restrict__ WoT,
    const float* __restrict__ bo, float* __restrict__ out) {
  __shared__ _Float16 As[2][128 * LSTR];
  __shared__ _Float16 Bs[2][128 * LSTR];
  const int tid = threadIdx.x, wave = tid >> 5, lane = tid & 31;
  const int nb = blockIdx.x * 128, mb = blockIdx.y * 128;
  const int mloc = (wave >> 1) * 32, nloc = (wave & 1) * 64;

  GemmAcc acc;
#pragma unroll
  for (int mi = 0; mi < 2; ++mi)
#pragma unroll
    for (int ni = 0; ni < 4; ++ni) { v8f z = {}; acc.a[mi][ni] = z; }

  gemm_core(ctx, WoT, As, Bs, mb, nb, DIM, tid, acc);

#pragma unroll
  for (int mi = 0; mi < 2; ++mi)
#pragma unroll
    for (int ni = 0; ni < 4; ++ni)
#pragma unroll
      for (int v = 0; v < 8; ++v) {
        int row = mb + mloc + mi * 16 + v + (lane >> 4) * 8;
        int col = nb + nloc + ni * 16 + (lane & 15);
        out[(size_t)row * DIM + col] = acc.a[mi][ni][v] + bo[col];
      }
}

// ---------------------------------------------------------------------------
extern "C" void kernel_launch(void* const* d_in, const int* in_sizes, int n_in,
                              void* d_out, int out_size, void* d_ws, size_t ws_size,
                              hipStream_t stream) {
  (void)in_sizes; (void)n_in; (void)out_size; (void)ws_size;
  const float* x    = (const float*)d_in[0];
  const float* Wqkv = (const float*)d_in[1];
  const float* bqkv = (const float*)d_in[2];
  const float* Wo   = (const float*)d_in[3];
  const float* bo   = (const float*)d_in[4];
  float* out = (float*)d_out;

  const size_t nh = (size_t)BATCH * HEADS * SEQ * HDIM;   // 8,388,608 halves
  _Float16* Q16    = (_Float16*)d_ws;                     // 16 MiB each
  _Float16* K16    = Q16 + nh;
  _Float16* Vt16   = K16 + nh;
  _Float16* ctx16  = Vt16 + nh;
  _Float16* x16    = ctx16 + nh;                          // [8192][1024]
  _Float16* WqkvT  = x16 + (size_t)MTOT * DIM;            // [3072][1024]
  _Float16* WoT    = WqkvT + (size_t)NQKV * DIM;          // [1024][1024]  (88 MiB total)

  cvt_f16_kernel<<<dim3((MTOT * DIM) / 256), 256, 0, stream>>>(x, x16);
  transpose_cvt_kernel<<<dim3(NQKV / 32, DIM / 32), 256, 0, stream>>>(Wqkv, WqkvT, DIM, NQKV);
  transpose_cvt_kernel<<<dim3(DIM / 32, DIM / 32), 256, 0, stream>>>(Wo, WoT, DIM, DIM);

  qkv_gemm_kernel<<<dim3(NQKV / 128, MTOT / 128), 256, 0, stream>>>(
      x16, WqkvT, bqkv, Q16, K16, Vt16);
  flash_attn_kernel<<<dim3(BATCH * HEADS * (SEQ / 128)), 256, 0, stream>>>(
      Q16, K16, Vt16, ctx16);
  out_gemm_kernel<<<dim3(DIM / 128, MTOT / 128), 256, 0, stream>>>(
      ctx16, WoT, bo, out);
}
